// GlobalPointer_66425964200076
// MI455X (gfx1250) — matrix operand order repaired and checked by reading, not compiled, not run on hardware
//
#include <hip/hip_runtime.h>
#include <hip/hip_bf16.h>

typedef __attribute__((ext_vector_type(16))) _Float16 v16h;
typedef __attribute__((ext_vector_type(8)))  _Float16 v8h;
typedef __attribute__((ext_vector_type(8)))  float    v8f;
typedef __attribute__((ext_vector_type(4)))  int      v4i;

#define HEADS 12
#define HDIM  64
#define DMODEL 768
#define NPROJ 1536   // HEADS * HDIM * 2
#define SEQ   1024
#define BATCH 8
#define NEGBIG 1000000000000.0f

typedef __attribute__((address_space(1))) v4i gv4i;   // global int4
typedef __attribute__((address_space(3))) v4i lv4i;   // LDS int4

// ---- fragment loaders (ISA 7.12.2 layouts, wave32) ----
// A 16x32 f16: lane m = lane&15; element e -> K = (e<8 ? e : e+8) + hi*8
__device__ inline v16h load_a_frag(const _Float16* row, int hi) {
    const v8h lo = *(const v8h*)(row + hi * 8);
    const v8h hp = *(const v8h*)(row + 16 + hi * 8);
    v16h a;
#pragma unroll
    for (int e = 0; e < 8; ++e) { a[e] = lo[e]; a[e + 8] = hp[e]; }
    return a;
}
// B 32x16 f16: lane n = lane&15; element e -> K = (e<8 ? e : e+8) + hi*8
__device__ inline v16h load_b_frag(const _Float16* row, int hi) {
    const v8h lo = *(const v8h*)(row + hi * 8);
    const v8h hp = *(const v8h*)(row + 16 + hi * 8);
    v16h b;
#pragma unroll
    for (int e = 0; e < 8; ++e) { b[e] = lo[e]; b[e + 8] = hp[e]; }
    return b;
}

// =====================================================================
// Kernel 1: proj = hidden @ W + b, RoPE, split -> Qh/Kh (f16)
// block = 256 thr (8 waves). Tile: 128 (M) x 64 (N). K-chunks of 32.
// =====================================================================
__global__ __launch_bounds__(256)
void proj_rope_kernel(const float* __restrict__ hs, const float* __restrict__ W,
                      const float* __restrict__ bias,
                      _Float16* __restrict__ Qh, _Float16* __restrict__ Kh) {
    __shared__ _Float16 As[128][40];  // 128 rows x 32 cols (+pad)
    __shared__ _Float16 Bs[64][40];   // 64 cols(n) x 32 rows(k), transposed (+pad)

    const int tid = threadIdx.x;
    const int l   = tid & 31;
    const int wid = tid >> 5;
    const int lm  = l & 15;
    const int hi  = l >> 4;
    const int waveM = wid >> 1;   // 0..3 -> 32 rows each
    const int waveN = wid & 1;    // 0..1 -> 32 cols each
    const int mBase = blockIdx.y * 128;
    const int nBase = blockIdx.x * 64;

    v8f acc[2][2] = {};

    for (int kc = 0; kc < DMODEL / 32; ++kc) {
        const int k0 = kc * 32;
        __syncthreads();
        // stage hidden tile (f32 -> f16): 128x32 = 1024 float4
#pragma unroll
        for (int i = 0; i < 4; ++i) {
            const int q = tid + i * 256;
            const int r = q >> 3, c4 = q & 7;
            const float* src = hs + (size_t)(mBase + r) * DMODEL + k0 + c4 * 4;
            if (kc + 1 < DMODEL / 32) __builtin_prefetch(src + 32, 0, 1);  // next k-chunk
            const float4 f = *(const float4*)src;
            As[r][c4 * 4 + 0] = (_Float16)f.x;
            As[r][c4 * 4 + 1] = (_Float16)f.y;
            As[r][c4 * 4 + 2] = (_Float16)f.z;
            As[r][c4 * 4 + 3] = (_Float16)f.w;
        }
        // stage W tile transposed (f32 -> f16): 32x64 = 512 float4
#pragma unroll
        for (int i = 0; i < 2; ++i) {
            const int q = tid + i * 256;
            const int kr = q >> 4, c4 = q & 15;
            const float* src = W + (size_t)(k0 + kr) * NPROJ + nBase + c4 * 4;
            if (kc + 1 < DMODEL / 32) __builtin_prefetch(src + 32 * NPROJ, 0, 1);
            const float4 w = *(const float4*)src;
            Bs[c4 * 4 + 0][kr] = (_Float16)w.x;
            Bs[c4 * 4 + 1][kr] = (_Float16)w.y;
            Bs[c4 * 4 + 2][kr] = (_Float16)w.z;
            Bs[c4 * 4 + 3][kr] = (_Float16)w.w;
        }
        __syncthreads();

        v16h af[2], bf[2];
#pragma unroll
        for (int tm = 0; tm < 2; ++tm)
            af[tm] = load_a_frag(&As[waveM * 32 + tm * 16 + lm][0], hi);
#pragma unroll
        for (int tn = 0; tn < 2; ++tn)
            bf[tn] = load_b_frag(&Bs[waveN * 32 + tn * 16 + lm][0], hi);
#pragma unroll
        for (int tm = 0; tm < 2; ++tm)
#pragma unroll
            for (int tn = 0; tn < 2; ++tn)
                acc[tm][tn] = __builtin_amdgcn_wmma_f32_16x16x32_f16(
                    false, af[tm], false, bf[tn], (short)0, acc[tm][tn], false, false);
    }

    // epilogue: bias + RoPE + scatter to Qh/Kh
#pragma unroll
    for (int tm = 0; tm < 2; ++tm) {
#pragma unroll
        for (int tn = 0; tn < 2; ++tn) {
            const int gn = nBase + waveN * 32 + tn * 16 + lm;     // proj column
            const float bv = bias[gn];
            const int d   = gn & 63;           // rot dim
            const int hh  = gn >> 7;           // head
            const int isk = (gn >> 6) & 1;     // q or k half
            const float invf = __powf(10000.0f, -(float)(d & ~1) / 64.0f);
            _Float16* dst = isk ? Kh : Qh;
            const v8f a = acc[tm][tn];
#pragma unroll
            for (int r = 0; r < 8; ++r) {
                const int gm = mBase + waveM * 32 + tm * 16 + r + hi * 8;
                const int t  = gm & (SEQ - 1);
                const float x  = a[r] + bv;
                const float xp = __shfl_xor(x, 1, 32);   // partner column (d^1)
                float s, c;
                __sincosf((float)t * invf, &s, &c);
                const float res = (d & 1) ? (x * c + xp * s) : (x * c - xp * s);
                dst[(size_t)gm * (HEADS * HDIM) + hh * HDIM + d] = (_Float16)res;
            }
        }
    }
}

// =====================================================================
// Kernel 2: logits[b,h,m,n] = (Q.Kt * mask - (1-mask)*NEG - tril*NEG)/8
// grid: x = n-tiles(16 of 64), y = m-tiles(8 of 128), z = b*12+h (96)
// Staging uses the CDNA5 async global->LDS path (ASYNCcnt).
// =====================================================================
__global__ __launch_bounds__(256)
void attn_logits_kernel(const _Float16* __restrict__ Qh, const _Float16* __restrict__ Kh,
                        const int* __restrict__ amask, float* __restrict__ out) {
    __shared__ _Float16 Qs[128][72];
    __shared__ _Float16 Ks[64][72];

    const int tid = threadIdx.x;
    const int l   = tid & 31;
    const int wid = tid >> 5;
    const int lm  = l & 15;
    const int hi  = l >> 4;
    const int waveM = wid >> 1;
    const int waveN = wid & 1;
    const int z  = blockIdx.z;
    const int b  = z / HEADS;
    const int hh = z % HEADS;
    const int mBase = blockIdx.y * 128;
    const int nBase = blockIdx.x * 64;

    // stage Q tile: 128 rows x 64 halves = 1024 x 16B (async global->LDS)
#pragma unroll
    for (int i = 0; i < 4; ++i) {
        const int q = tid + i * 256;
        const int r = q >> 3, c = (q & 7) * 8;
        const _Float16* gsrc =
            Qh + (size_t)(b * SEQ + mBase + r) * (HEADS * HDIM) + hh * HDIM + c;
        __builtin_amdgcn_global_load_async_to_lds_b128(
            (gv4i*)gsrc, (lv4i*)&Qs[r][c], 0, 0);
    }
    // stage K tile: 64 rows x 64 halves = 512 x 16B (async global->LDS)
#pragma unroll
    for (int i = 0; i < 2; ++i) {
        const int q = tid + i * 256;
        const int r = q >> 3, c = (q & 7) * 8;
        const _Float16* gsrc =
            Kh + (size_t)(b * SEQ + nBase + r) * (HEADS * HDIM) + hh * HDIM + c;
        __builtin_amdgcn_global_load_async_to_lds_b128(
            (gv4i*)gsrc, (lv4i*)&Ks[r][c], 0, 0);
    }
    __builtin_amdgcn_s_wait_asynccnt(0);   // this wave's async copies done
    __syncthreads();                        // all waves' copies visible

    v8f acc[2][2] = {};
#pragma unroll
    for (int chunk = 0; chunk < 2; ++chunk) {
        const int c0 = chunk * 32;
        v16h af[2], bf[2];
#pragma unroll
        for (int tm = 0; tm < 2; ++tm)
            af[tm] = load_a_frag(&Qs[waveM * 32 + tm * 16 + lm][c0], hi);
#pragma unroll
        for (int tn = 0; tn < 2; ++tn)
            bf[tn] = load_b_frag(&Ks[waveN * 32 + tn * 16 + lm][c0], hi);
#pragma unroll
        for (int tm = 0; tm < 2; ++tm)
#pragma unroll
            for (int tn = 0; tn < 2; ++tn)
                acc[tm][tn] = __builtin_amdgcn_wmma_f32_16x16x32_f16(
                    false, af[tm], false, bf[tn], (short)0, acc[tm][tn], false, false);
    }

    // epilogue: mask + strict-lower-triangle bias + scale; stream out (NT)
#pragma unroll
    for (int tn = 0; tn < 2; ++tn) {
        const int gn = nBase + waveN * 32 + tn * 16 + lm;       // key position
        const float mk  = (float)amask[b * SEQ + gn];
        const float sub = (1.0f - mk) * NEGBIG;
#pragma unroll
        for (int tm = 0; tm < 2; ++tm) {
            const v8f a = acc[tm][tn];
#pragma unroll
            for (int r = 0; r < 8; ++r) {
                const int gm = mBase + waveM * 32 + tm * 16 + r + hi * 8;  // query position
                float v = a[r] * mk - sub - (gn < gm ? NEGBIG : 0.0f);
                v *= 0.125f;  // 1/sqrt(64)
                __builtin_nontemporal_store(v, out + ((size_t)z * SEQ + gm) * SEQ + gn);
            }
        }
    }
}

extern "C" void kernel_launch(void* const* d_in, const int* in_sizes, int n_in,
                              void* d_out, int out_size, void* d_ws, size_t ws_size,
                              hipStream_t stream) {
    const float* hs   = (const float*)d_in[0];   // (8,1024,768)
    const int*   am   = (const int*)d_in[1];     // (8,1024)
    const float* W    = (const float*)d_in[2];   // (768,1536)
    const float* bias = (const float*)d_in[3];   // (1536,)
    float* out = (float*)d_out;                  // (8,12,1024,1024)

    const size_t qk_elems = (size_t)BATCH * SEQ * HEADS * HDIM;  // 6.29M halves each
    _Float16* Qh = (_Float16*)d_ws;
    _Float16* Kh = Qh + qk_elems;

    dim3 grid1(NPROJ / 64, (BATCH * SEQ) / 128);    // (24, 64)
    proj_rope_kernel<<<grid1, 256, 0, stream>>>(hs, W, bias, Qh, Kh);

    dim3 grid2(SEQ / 64, SEQ / 128, BATCH * HEADS); // (16, 8, 96)
    attn_logits_kernel<<<grid2, 256, 0, stream>>>(Qh, Kh, am, out);
}
